// SDPA_TRITON_FA_3624952397975
// MI455X (gfx1250) — matrix operand order repaired and checked by reading, not compiled, not run on hardware
//
#include <hip/hip_runtime.h>
#include <hip/hip_bf16.h>

// -------- problem constants (match reference) --------
#define BATCH   2
#define HEADS   16
#define SEQLEN  2048
#define DHEAD   64
// softmax scale folded with log2(e) so we can use exp2 throughout
#define QK_SCALE (0.125f * 1.4426950408889634f)

// -------- tiling --------
#define BLOCK_M 128    // 8 waves * 16 rows each
#define BLOCK_N 64     // KV chunk (2 WMMA K-groups of 32)
#define NWAVES  8
#define NTHREADS 256
#define DV      (DHEAD + 16)   // V extended with 16 "ones" columns for row-sum

typedef __attribute__((ext_vector_type(16))) _Float16 v16h;
typedef __attribute__((ext_vector_type(8)))  _Float16 v8h;
typedef __attribute__((ext_vector_type(4)))  _Float16 v4h;
typedef __attribute__((ext_vector_type(2)))  _Float16 v2h;
typedef __attribute__((ext_vector_type(8)))  float    v8f;
typedef __attribute__((ext_vector_type(4)))  float    v4f;

// packed f32->2xf16 convert (builtin returns __fp16 vector; bit-cast to v2h)
__device__ __forceinline__ v2h cvt_pk(float a, float b) {
    return __builtin_bit_cast(v2h, __builtin_amdgcn_cvt_pkrtz(a, b));
}

// ---- canonicalize-free max ops (asm is opaque to llvm.maxnum legalization) ----
__device__ __forceinline__ float vmaxf(float a, float b) {
    float d;
    asm("v_max_num_f32 %0, %1, %2" : "=v"(d) : "v"(a), "v"(b));
    return d;
}
__device__ __forceinline__ v2h pkmax(v2h a, v2h b) {
    v2h d;
    asm("v_pk_max_num_f16 %0, %1, %2" : "=v"(d) : "v"(a), "v"(b));
    return d;
}

// 16-lane XOR-butterfly max on packed f16 pairs (f16 max is exact-enough:
// softmax is shift-invariant and corr errors cancel in the O/l ratio).
__device__ __forceinline__ v2h pkrowmax16(v2h x) {
#define STEP(IMM) \
    x = pkmax(x, __builtin_bit_cast(v2h, \
            __builtin_amdgcn_ds_swizzle(__builtin_bit_cast(int, x), IMM)))
    STEP(0x041f); STEP(0x081f); STEP(0x101f); STEP(0x201f);
#undef STEP
    return x;
}

__device__ __forceinline__ v16h ld_a16(const _Float16* p_lo, const _Float16* p_hi) {
    v8h lo = *(const v8h*)p_lo;
    v8h hi = *(const v8h*)p_hi;
    v16h a;
#pragma unroll
    for (int e = 0; e < 8; ++e) { a[e] = lo[e]; a[e + 8] = hi[e]; }
    return a;
}

// cooperative K/V chunk staging (f16; V transposed + K-permuted, per 32-group:
// stream position s of group g holds kv = g*32 + (s>>1) + 16*(s&1))
__device__ __forceinline__ void stage_kv(const float* __restrict__ kb,
                                         const float* __restrict__ vb,
                                         int kv0, _Float16* ks, _Float16* vt,
                                         int tid) {
    // ---- K chunk: row-major [n][d], 64x64; 16 floats per thread ----
    {
        const float* src = kb + kv0 * DHEAD + tid * 16;
        v4f a0 = *(const v4f*)(src);
        v4f a1 = *(const v4f*)(src + 4);
        v4f a2 = *(const v4f*)(src + 8);
        v4f a3 = *(const v4f*)(src + 12);
        v8h h;
        v2h t0 = cvt_pk(a0[0], a0[1]), t1 = cvt_pk(a0[2], a0[3]);
        v2h t2 = cvt_pk(a1[0], a1[1]), t3 = cvt_pk(a1[2], a1[3]);
        h[0]=t0[0]; h[1]=t0[1]; h[2]=t1[0]; h[3]=t1[1];
        h[4]=t2[0]; h[5]=t2[1]; h[6]=t3[0]; h[7]=t3[1];
        *(v8h*)(ks + tid * 16) = h;
        t0 = cvt_pk(a2[0], a2[1]); t1 = cvt_pk(a2[2], a2[3]);
        t2 = cvt_pk(a3[0], a3[1]); t3 = cvt_pk(a3[2], a3[3]);
        h[0]=t0[0]; h[1]=t0[1]; h[2]=t1[0]; h[3]=t1[1];
        h[4]=t2[0]; h[5]=t2[1]; h[6]=t3[0]; h[7]=t3[1];
        *(v8h*)(ks + tid * 16 + 8) = h;
    }
    // ---- V chunk: 32 pair-units (16 per group), 8 threads per unit ----
    {
        const int unit = tid >> 3;           // 0..31
        const int g    = unit >> 4;          // K-group
        const int kvr  = unit & 15;          // pair row within group
        const int d0   = (tid & 7) * 8;      // 8 d-values per thread
        const float* r0 = vb + (kv0 + g * 32 + kvr) * DHEAD + d0;
        const float* r1 = r0 + 16 * DHEAD;   // partner row kv+16
        v4f a0 = *(const v4f*)(r0);
        v4f a1 = *(const v4f*)(r0 + 4);
        v4f b0 = *(const v4f*)(r1);
        v4f b1 = *(const v4f*)(r1 + 4);
        const int kp = g * 32 + kvr * 2;     // packed K-position
#pragma unroll
        for (int e = 0; e < 4; ++e) {
            *(v2h*)(vt + (d0 + e) * BLOCK_N + kp)     = cvt_pk(a0[e], b0[e]);
            *(v2h*)(vt + (d0 + 4 + e) * BLOCK_N + kp) = cvt_pk(a1[e], b1[e]);
        }
    }
}

__global__ __launch_bounds__(NTHREADS)
void fa_fwd_wmma_f16(const float* __restrict__ q,
                     const float* __restrict__ k,
                     const float* __restrict__ v,
                     float* __restrict__ out) {
    __shared__ _Float16 Ks[2][BLOCK_N * DHEAD];       // double-buffered, 8KB each
    __shared__ _Float16 Vt[2][DV * BLOCK_N];          // double-buffered, 10KB each
    __shared__ _Float16 Qs[NWAVES * 16 * DHEAD];      // 16KB
    __shared__ _Float16 Ps[NWAVES * 16 * BLOCK_N];    // 16KB

    const int tid   = threadIdx.x;
    const int wave  = tid >> 5;
    const int lane  = tid & 31;
    const int laneN = lane & 15;
    const int hi    = lane >> 4;          // 0: lanes 0-15 (rows 0-7), 1: rows 8-15

    const int bh = blockIdx.y;
    const int m0 = blockIdx.x * BLOCK_M + wave * 16;

    const float* qb = q + ((size_t)bh * SEQLEN + m0) * DHEAD;
    const float* kb = k + (size_t)bh * SEQLEN * DHEAD;
    const float* vb = v + (size_t)bh * SEQLEN * DHEAD;
    float*       ob = out + ((size_t)bh * SEQLEN + m0) * DHEAD;

    // ---- ones-extension of V (rows d=64..79, full K width) in both buffers ----
    {
        v4h one4;
#pragma unroll
        for (int e = 0; e < 4; ++e) one4[e] = (_Float16)1.0f;
        *(v4h*)(Vt[0] + DHEAD * BLOCK_N + tid * 4) = one4;   // 1024 halves
        *(v4h*)(Vt[1] + DHEAD * BLOCK_N + tid * 4) = one4;
    }

    // ---- stage Q tile (16x64) to LDS as f16, scale folded in ----
    _Float16* qsm = Qs + wave * 16 * DHEAD;
#pragma unroll
    for (int j = 0; j < 8; ++j) {
        const int f4 = j * 32 + lane;
        v4f qv = *(const v4f*)(qb + f4 * 4);
        v2h q0 = cvt_pk(qv[0] * QK_SCALE, qv[1] * QK_SCALE);
        v2h q1 = cvt_pk(qv[2] * QK_SCALE, qv[3] * QK_SCALE);
        v4h qh; qh[0]=q0[0]; qh[1]=q0[1]; qh[2]=q1[0]; qh[3]=q1[1];
        *(v4h*)(qsm + f4 * 4) = qh;
    }
    asm volatile("s_wait_dscnt 0" ::: "memory");

    v16h aq[2];
#pragma unroll
    for (int kc = 0; kc < 2; ++kc) {
        const _Float16* base = qsm + laneN * DHEAD + kc * 32 + hi * 16;
        aq[kc] = ld_a16(base, base + 8);
    }

    // ---- online-softmax state (row-sum lives in acc[4] via ones-columns) ----
    v8f acc[5] = {};
    float m_i[8];
#pragma unroll
    for (int r = 0; r < 8; ++r) m_i[r] = -1.0e30f;

    _Float16* psm = Ps + wave * 16 * BLOCK_N;

    // preload chunk 0
    stage_kv(kb, vb, 0, Ks[0], Vt[0], tid);
    __syncthreads();

    const int NCHUNK = SEQLEN / BLOCK_N;
    for (int i = 0; i < NCHUNK; ++i) {
        const int cur = i & 1;
        if (i + 1 < NCHUNK) {    // stage next chunk; overlaps with WMMA below
            stage_kv(kb, vb, (i + 1) * BLOCK_N, Ks[cur ^ 1], Vt[cur ^ 1], tid);
            __builtin_prefetch(kb + (i + 2) * BLOCK_N * DHEAD + tid * 16, 0, 0);
            __builtin_prefetch(vb + (i + 2) * BLOCK_N * DHEAD + tid * 16, 0, 0);
        }
        const _Float16* ks = Ks[cur];
        const _Float16* vt = Vt[cur];

        // ---- S = Q * K^T  (16x64 scores = four 16x16 C tiles) ----
        v8f c[4] = {};
#pragma unroll
        for (int kc = 0; kc < 2; ++kc) {
            const int kk0 = kc * 32 + hi * 16;
#pragma unroll
            for (int j = 0; j < 4; ++j) {
                v16h b = *(const v16h*)(ks + (j * 16 + laneN) * DHEAD + kk0);
                c[j] = __builtin_amdgcn_wmma_f32_16x16x32_f16(false, aq[kc], false, b,
                                                              (short)0, c[j], false, false);
            }
        }

        // ---- packed-f16 row-max butterfly (rows pair-packed: 4 regs) ----
        v2h pm[4];
#pragma unroll
        for (int r2 = 0; r2 < 4; ++r2) {
            const int r0 = 2 * r2, r1 = 2 * r2 + 1;
            pm[r2] = cvt_pk(vmaxf(vmaxf(c[0][r0], c[1][r0]), vmaxf(c[2][r0], c[3][r0])),
                            vmaxf(vmaxf(c[0][r1], c[1][r1]), vmaxf(c[2][r1], c[3][r1])));
            pm[r2] = pkrowmax16(pm[r2]);
        }

        // ---- online softmax update; sum comes free via acc[4] ----
#pragma unroll
        for (int r = 0; r < 8; ++r) {
            const float cm   = (float)pm[r / 2][r & 1];
            const float mnew = vmaxf(m_i[r], cm);
            const float corr = __builtin_amdgcn_exp2f(m_i[r] - mnew);
            m_i[r] = mnew;
            const float p0 = __builtin_amdgcn_exp2f(c[0][r] - mnew);
            const float p1 = __builtin_amdgcn_exp2f(c[1][r] - mnew);
            const float p2 = __builtin_amdgcn_exp2f(c[2][r] - mnew);
            const float p3 = __builtin_amdgcn_exp2f(c[3][r] - mnew);
#pragma unroll
            for (int j = 0; j < 5; ++j) acc[j][r] *= corr;
            // packed P spill: (p0,p1) -> group0, (p2,p3) -> group1
            _Float16* prow = psm + (r + hi * 8) * BLOCK_N;
            *(v2h*)(prow + laneN * 2)      = cvt_pk(p0, p1);
            *(v2h*)(prow + 32 + laneN * 2) = cvt_pk(p2, p3);
        }
        asm volatile("s_wait_dscnt 0" ::: "memory");

        // ---- reload P as two WMMA A-operands (16x32 each, K-permuted) ----
        v16h ap[2];
#pragma unroll
        for (int g = 0; g < 2; ++g) {
            const _Float16* pbase = psm + laneN * BLOCK_N + g * 32 + hi * 16;
            ap[g] = ld_a16(pbase, pbase + 8);
        }

        // ---- O += P * V  (4 output tiles + 1 row-sum tile, K=64) ----
#pragma unroll
        for (int j = 0; j < 5; ++j) {
#pragma unroll
            for (int g = 0; g < 2; ++g) {
                v16h bv = *(const v16h*)(vt + (j * 16 + laneN) * BLOCK_N + g * 32 + hi * 16);
                acc[j] = __builtin_amdgcn_wmma_f32_16x16x32_f16(false, ap[g], false, bv,
                                                                (short)0, acc[j], false, false);
            }
        }
        __syncthreads();   // next-chunk staging complete; cur free for reuse
    }

    // ---- epilogue: normalize by l (= acc[4], replicated in every lane) ----
#pragma unroll
    for (int r = 0; r < 8; ++r) {
        const float invl = 1.0f / acc[4][r];
        const int row = r + hi * 8;
#pragma unroll
        for (int j = 0; j < 4; ++j)
            ob[row * DHEAD + j * 16 + laneN] = acc[j][r] * invl;
    }
}

extern "C" void kernel_launch(void* const* d_in, const int* in_sizes, int n_in,
                              void* d_out, int out_size, void* d_ws, size_t ws_size,
                              hipStream_t stream) {
    (void)in_sizes; (void)n_in; (void)d_ws; (void)ws_size; (void)out_size;
    const float* q = (const float*)d_in[0];
    const float* k = (const float*)d_in[1];
    const float* v = (const float*)d_in[2];
    float* out = (float*)d_out;

    dim3 grid(SEQLEN / BLOCK_M, BATCH * HEADS, 1);
    dim3 block(NTHREADS, 1, 1);
    fa_fwd_wmma_f16<<<grid, block, 0, stream>>>(q, k, v, out);
}
